// HNN_30185030157095
// MI455X (gfx1250) — matrix operand order repaired and checked by reading, not compiled, not run on hardware
//
#include <hip/hip_runtime.h>
#include <stdint.h>

typedef __attribute__((ext_vector_type(16))) _Float16 v16h;
typedef __attribute__((ext_vector_type(8)))  _Float16 v8h;
typedef __attribute__((ext_vector_type(2)))  _Float16 v2h;
typedef __attribute__((ext_vector_type(8)))  float    v8f;

#define HIDDEN     128
#define LDSH       136        // padded row stride in halves: 272B = 16B-aligned, bank-skewed
#define ZDIM       18
#define EPS2_C     0.01f
#define MASS_EPS_C 1e-8f
#define WAVES      8
#define BPW        5          // batches per wave (15 real rows + 1 dummy = M=16)
#define BPWG       (WAVES*BPW)
#define ITERS      4          // row-slabs per wave -> 160 batches per WG

// Prebuilt weight image (built once in d_ws by hnn_prep, async-copied to LDS).
struct WImage {
  _Float16 w2 [HIDDEN * LDSH];   // [n][k] = W2[n][k]   (forward  B frags)
  _Float16 w2t[HIDDEN * LDSH];   // [n][k] = W2[k][n]   (backward B frags)
  float w1[HIDDEN], b1v[HIDDEN], b2v[HIDDEN], w3[HIDDEN];
};
// sizeof(WImage) = 2*34816 + 2048 = 71680 bytes, multiple of 16.

__device__ __forceinline__ void lds_fence() {
  asm volatile("s_wait_dscnt 0" ::: "memory");
}

__device__ __forceinline__ float sigmoid_fast(float x) {
  // v_exp_f32 + v_rcp_f32 (1-ulp) instead of IEEE divide sequence
  return __builtin_amdgcn_rcpf(1.0f + __expf(-x));
}

__device__ __forceinline__ v16h load16(const _Float16* p0, const _Float16* p1) {
  v8h lo = *(const v8h*)p0;
  v8h hi = *(const v8h*)p1;
  v16h r;
#pragma unroll
  for (int i = 0; i < 8; ++i) { r[i] = lo[i]; r[i + 8] = hi[i]; }
  return r;
}

__global__ __launch_bounds__(256) void hnn_prep(
    const float* __restrict__ W1, const float* __restrict__ b1,
    const float* __restrict__ W2, const float* __restrict__ b2,
    const float* __restrict__ W3, WImage* __restrict__ img)
{
  const int stride = gridDim.x * 256;
  for (int idx = blockIdx.x * 256 + threadIdx.x; idx < HIDDEN * HIDDEN; idx += stride) {
    int r = idx >> 7, c = idx & (HIDDEN - 1);
    _Float16 w = (_Float16)W2[idx];
    img->w2 [r * LDSH + c] = w;
    img->w2t[c * LDSH + r] = w;
  }
  if (blockIdx.x == 0) {
    for (int n = threadIdx.x; n < HIDDEN; n += 256) {
      img->w1[n] = W1[n]; img->b1v[n] = b1[n];
      img->b2v[n] = b2[n]; img->w3[n] = W3[n];
    }
  }
}

__global__ __launch_bounds__(256) void hnn_grad_kernel(
    const float* __restrict__ z, const float* __restrict__ logm,
    const WImage* __restrict__ img, float* __restrict__ out, int nBatch)
{
  __shared__ alignas(16) WImage sImg;
  __shared__ alignas(16) _Float16 sStage[WAVES][16 * LDSH];  // h1, then da2 (reused)
  __shared__ float sGeom[WAVES][16][4];   // diff.xyz, 1/d per row
  __shared__ float sDs[WAVES][16];        // dv/ds per row

  const int tid = threadIdx.x;

  // ---- preamble: async-copy prebuilt weight image into LDS ----
  {
    const char* gsrc = (const char*)img;
    uint32_t lbase = (uint32_t)(uintptr_t)(void*)&sImg;   // low 32 bits = LDS address
    for (int t = tid; t < (int)(sizeof(WImage) / 16); t += 256) {
      uint32_t la = lbase + (uint32_t)t * 16u;
      const char* ga = gsrc + (size_t)t * 16u;
      asm volatile("global_load_async_to_lds_b128 %0, %1, off"
                   :: "v"(la), "v"(ga) : "memory");
    }
    asm volatile("s_wait_asynccnt 0" ::: "memory");
  }
  __syncthreads();

  const int wave = tid >> 5;
  const int lane = tid & 31;
  const int hs   = lane >> 4;     // half-select (0: lanes 0-15, 1: lanes 16-31)
  const int lm   = lane & 15;
  _Float16* stage = sStage[wave];

  float minv[3];
#pragma unroll
  for (int k = 0; k < 3; ++k) minv[k] = 1.0f / (__expf(logm[k]) + MASS_EPS_C);

  for (int it = 0; it < ITERS; ++it) {
    const long base = ((long)blockIdx.x * ITERS + it) * BPWG + (long)wave * BPW;
    if (base >= (long)nBatch) break;
    lds_fence();   // previous-iteration sGeom/sDs readers are done (in-order LDS)

    // ---- geometry: s = 1/d for the wave's 16 rows (row 15 = dummy) ----
    long bRowL = base + (lm < 15 ? lm / 3 : 0);
    if (bRowL > (long)(nBatch - 1)) bRowL = nBatch - 1;
    const int bRow = (int)bRowL;
    const int pRow = (lm < 15) ? (lm % 3) : 0;
    const int bi = (pRow == 2) ? 1 : 0;   // PAIR_I = {0,0,1}
    const int bj = (pRow == 0) ? 1 : 2;   // PAIR_J = {1,2,2}
    const float* zr0 = z + (size_t)bRow * ZDIM;
    float dfx = zr0[bi*3+0] - zr0[bj*3+0];
    float dfy = zr0[bi*3+1] - zr0[bj*3+1];
    float dfz = zr0[bi*3+2] - zr0[bj*3+2];
    float d2   = dfx*dfx + dfy*dfy + dfz*dfz + EPS2_C;
    float dinv = __builtin_amdgcn_rsqf(d2);
    float sval = dinv;                    // s = 1/d
    if (lane < 16) {
      sGeom[wave][lm][0] = dfx; sGeom[wave][lm][1] = dfy;
      sGeom[wave][lm][2] = dfz; sGeom[wave][lm][3] = dinv;
    }

    // ---- layer 1: h1 = silu(W1*s + b1), staged 16x128 f16 (packed stores) ----
#pragma unroll
    for (int mm = 0; mm < 16; ++mm) {
      float sm = __shfl(sval, mm, 32);
#pragma unroll
      for (int nb = 0; nb < 2; ++nb) {
        int n0 = nb * 64 + lane * 2;
        float z1a = sImg.w1[n0]     * sm + sImg.b1v[n0];
        float z1b = sImg.w1[n0 + 1] * sm + sImg.b1v[n0 + 1];
        v2h pk;
        pk[0] = (_Float16)(z1a * sigmoid_fast(z1a));
        pk[1] = (_Float16)(z1b * sigmoid_fast(z1b));
        *(v2h*)&stage[mm * LDSH + n0] = pk;
      }
    }
    lds_fence();

    // ---- layer 2 forward: a2(16x128) = h1 @ W2^T  (8 N-tiles x 4 K-tiles) ----
    v16h afrag[4];
#pragma unroll
    for (int kt = 0; kt < 4; ++kt) {
      int k0 = kt * 32 + hs * 8;          // A layout: K {0-7,16-23} / {8-15,24-31}
      afrag[kt] = load16(&stage[lm * LDSH + k0], &stage[lm * LDSH + k0 + 16]);
    }
    v8f acc[8];
#pragma unroll
    for (int nt = 0; nt < 8; ++nt) {
      v8f c = {0.f,0.f,0.f,0.f,0.f,0.f,0.f,0.f};
#pragma unroll
      for (int kt = 0; kt < 4; ++kt) {
        int n  = nt * 16 + lm;
        int k0 = kt * 32 + hs * 8;        // B layout: K {0-7,16-23} / {8-15,24-31}
        v16h bf = load16(&sImg.w2[n * LDSH + k0], &sImg.w2[n * LDSH + k0 + 16]);
        c = __builtin_amdgcn_wmma_f32_16x16x32_f16(false, afrag[kt], false, bf,
                                                   (short)0, c, false, false);
      }
      acc[nt] = c;
    }

    // ---- seed backward: da2 = W3 * silu'(a2 + b2), restage 16x128 f16 ----
#pragma unroll
    for (int nt = 0; nt < 8; ++nt) {
      int n = nt * 16 + lm;
      float w3 = sImg.w3[n], bb2 = sImg.b2v[n];
#pragma unroll
      for (int v = 0; v < 8; ++v) {
        float z2 = acc[nt][v] + bb2;
        float sg = sigmoid_fast(z2);
        float da = w3 * sg * (1.0f + z2 * (1.0f - sg));
        int mr = v + hs * 8;              // C layout: M = v + 8*half
        stage[mr * LDSH + n] = (_Float16)da;
      }
    }
    lds_fence();

    // ---- layer 2 backward: dh1(16x128) = da2 @ W2, fused with layer-1 bwd ----
    v16h gfrag[4];
#pragma unroll
    for (int kt = 0; kt < 4; ++kt) {
      int k0 = kt * 32 + hs * 8;
      gfrag[kt] = load16(&stage[lm * LDSH + k0], &stage[lm * LDSH + k0 + 16]);
    }
    float smv[8];
#pragma unroll
    for (int v = 0; v < 8; ++v) smv[v] = __shfl(sval, v + hs * 8, 32);

    float dsp[8] = {0,0,0,0,0,0,0,0};
#pragma unroll
    for (int nt = 0; nt < 8; ++nt) {
      v8f c = {0.f,0.f,0.f,0.f,0.f,0.f,0.f,0.f};
#pragma unroll
      for (int kt = 0; kt < 4; ++kt) {
        int n  = nt * 16 + lm;
        int k0 = kt * 32 + hs * 8;
        v16h bf = load16(&sImg.w2t[n * LDSH + k0], &sImg.w2t[n * LDSH + k0 + 16]);
        c = __builtin_amdgcn_wmma_f32_16x16x32_f16(false, gfrag[kt], false, bf,
                                                   (short)0, c, false, false);
      }
      int n = nt * 16 + lm;
      float w1 = sImg.w1[n], bb1 = sImg.b1v[n];
#pragma unroll
      for (int v = 0; v < 8; ++v) {
        float z1 = w1 * smv[v] + bb1;
        float sg = sigmoid_fast(z1);
        float dsilu = sg * (1.0f + z1 * (1.0f - sg));
        dsp[v] += c[v] * dsilu * w1;      // dv/ds partial for row (v + 8*hs)
      }
    }

    // ---- reduce dv/ds across the 16 lanes of each half ----
#pragma unroll
    for (int v = 0; v < 8; ++v) {
      float t = dsp[v];
      t += __shfl_xor(t, 1, 32);
      t += __shfl_xor(t, 2, 32);
      t += __shfl_xor(t, 4, 32);
      t += __shfl_xor(t, 8, 32);
      dsp[v] = t;
    }
    if (lm == 0) {                        // lanes 0 and 16 hold the totals
#pragma unroll
      for (int v = 0; v < 8; ++v) sDs[wave][hs * 8 + v] = dsp[v];
    }
    lds_fence();

    // ---- final per-batch assembly (5 lanes, one batch each) ----
    if (lane < BPW) {
      long bb = base + lane;
      if (bb < (long)nBatch) {
        float accq[9] = {0,0,0,0,0,0,0,0,0};
#pragma unroll
        for (int p = 0; p < 3; ++p) {
          int m = lane * 3 + p;
          float dsv = sDs[wave][m];
          float gx = sGeom[wave][m][0], gy = sGeom[wave][m][1];
          float gz = sGeom[wave][m][2], di = sGeom[wave][m][3];
          float g = dsv * di * di * di;   // dpdt_i = +ds*diff/d^3, dpdt_j = -...
          int i = (p == 2) ? 1 : 0;
          int j = (p == 0) ? 1 : 2;
          accq[3*i+0] += g * gx; accq[3*i+1] += g * gy; accq[3*i+2] += g * gz;
          accq[3*j+0] -= g * gx; accq[3*j+1] -= g * gy; accq[3*j+2] -= g * gz;
        }
        const float* zp = z + (size_t)bb * ZDIM;
        float* op = out + (size_t)bb * ZDIM;
#pragma unroll
        for (int k = 0; k < 9; ++k) {
          op[k]     = zp[9 + k] * minv[k / 3];  // dqdt = p * m^-1
          op[9 + k] = accq[k];                  // dpdt
        }
      }
    }
  }
}

extern "C" void kernel_launch(void* const* d_in, const int* in_sizes, int n_in,
                              void* d_out, int out_size, void* d_ws, size_t ws_size,
                              hipStream_t stream) {
  const float* z   = (const float*)d_in[0];
  const float* lm  = (const float*)d_in[1];
  const float* W1  = (const float*)d_in[2];
  const float* b1  = (const float*)d_in[3];
  const float* W2  = (const float*)d_in[4];
  const float* b2  = (const float*)d_in[5];
  const float* W3  = (const float*)d_in[6];
  float* out = (float*)d_out;
  (void)n_in; (void)out_size; (void)ws_size; (void)d_in;

  WImage* img = (WImage*)d_ws;
  int nBatch = in_sizes[0] / ZDIM;                        // 262144
  hipLaunchKernelGGL(hnn_prep, dim3(32), dim3(256), 0, stream,
                     W1, b1, W2, b2, W3, img);
  int grid = (nBatch + BPWG * ITERS - 1) / (BPWG * ITERS); // 1639
  hipLaunchKernelGGL(hnn_grad_kernel, dim3(grid), dim3(256), 0, stream,
                     z, lm, img, out, nBatch);
}
